// CARAFE_71201967833631
// MI455X (gfx1250) — compile-verified, matrix operand order checked by reading, not run on hardware
//
#include <hip/hip_runtime.h>
#include <hip/hip_bf16.h>

// ---------------------------------------------------------------------------
// CARAFE fused, round 4: branch-free epilogues (pre-zeroed y1 + address
// redirect), vector bias loads. All matrix math on v_wmma_f32_16x16x32_bf16.
// ---------------------------------------------------------------------------

typedef __attribute__((ext_vector_type(16))) __bf16 v16bf;
typedef __attribute__((ext_vector_type(8)))  __bf16 v8bf;
typedef __attribute__((ext_vector_type(2)))  __bf16 v2bf;
typedef __attribute__((ext_vector_type(8)))  float  v8f;
typedef __attribute__((ext_vector_type(4)))  float  v4f;
typedef __attribute__((ext_vector_type(2)))  float  v2f;

#define SF    2
#define KUP   5
#define MCH   48
#define CIN   64
#define HH    128
#define WWID  128
#define NB    8
#define OENC  100
#define OPAD  112     // 7*16 (12 pad outputs -> row slots 100..115)
#define KENC  432     // 9 taps * 48, tap-major
#define KENCP 448     // 14 * 32
#define TILE  16
#define H1    18
#define P1    324
#define H2    20
#define P2    400
#define XTS   72      // xt row stride (bf16): 144B, 16B aligned
#define Y1S   48      // y1t row stride (bf16): 96B
#define Y2S   116     // y2t row stride (f32): 464B, 16B aligned
                      // row layout: [k*4+s] for k<25,s<4 ; 100..115 scratch

constexpr int OFF_XT = 0;                          // bf16 [400][72]  =  57600 B
constexpr int OFF_WC = OFF_XT + P2 * XTS * 2;      // bf16 [48][64]   =   6144 B
constexpr int OFF_Y1 = OFF_WC + MCH * CIN * 2;     // bf16 [325][48]  =  31200 B
constexpr int OFF_WE = OFF_Y1 + 325 * Y1S * 2;     // bf16 [112][448] = 100352 B
constexpr int OFF_Y2 = OFF_WE + OPAD * KENCP * 2;  // f32  [256][116] = 118784 B
constexpr int OFF_BC = OFF_Y2 + 256 * Y2S * 4;     // f32  [48]
constexpr int OFF_BE = OFF_BC + MCH * 4;           // f32  [112] (zero padded)
constexpr int SMEM_BYTES = OFF_BE + OPAD * 4;      // 314,720 B < 320 KB

__device__ __forceinline__ v16bf cat16(v8bf lo, v8bf hi) {
    return __builtin_shufflevector(lo, hi, 0, 1, 2, 3, 4, 5, 6, 7,
                                          8, 9, 10, 11, 12, 13, 14, 15);
}

__global__ __launch_bounds__(256, 1)
void carafe_fused(const float* __restrict__ x,
                  const float* __restrict__ wcmp,
                  const float* __restrict__ bcmp,
                  const float* __restrict__ wenc_g,
                  const float* __restrict__ benc_g,
                  float* __restrict__ out)
{
    extern __shared__ char smem[];
    __bf16* xt  = (__bf16*)(smem + OFF_XT);   // [pos(400)][chan(72)]
    __bf16* wc  = (__bf16*)(smem + OFF_WC);   // [m(48)][c(64)]
    __bf16* y1t = (__bf16*)(smem + OFF_Y1);   // [pos(324)+dummy][m(48)]
    __bf16* we  = (__bf16*)(smem + OFF_WE);   // [o(112)][k=tap*48+m (448)]
    float*  y2t = (float*) (smem + OFF_Y2);   // [pos(256)][k*4+s (116)]
    float*  bc  = (float*) (smem + OFF_BC);
    float*  be  = (float*) (smem + OFF_BE);

    const int tid  = threadIdx.x;
    const int lane = tid & 31;
    const int wv   = tid >> 5;                // 8 wave32 / block
    const int b    = blockIdx.z;
    const int th0  = blockIdx.y * TILE;
    const int tw0  = blockIdx.x * TILE;

    // ---------- Phase 0: stage x tile (halo 2, transposed, 2ch packed)
    const float* xb = x + (size_t)b * CIN * HH * WWID;
    for (int idx = tid; idx < (CIN / 2) * P2; idx += 256) {
        int c2 = idx / P2, pp = idx - c2 * P2;
        int gr = th0 + pp / H2 - 2;
        int gc = tw0 + pp % H2 - 2;
        int c  = c2 * 2;
        float v0 = 0.f, v1 = 0.f;
        if ((unsigned)gr < (unsigned)HH && (unsigned)gc < (unsigned)WWID) {
            const float* p = &xb[(c * HH + gr) * WWID + gc];
            __builtin_prefetch(p + WWID, 0, 1);       // global_prefetch_b8
            v0 = p[0];
            v1 = p[HH * WWID];
        }
        v2bf t; t[0] = (__bf16)v0; t[1] = (__bf16)v1;
        *(v2bf*)(xt + pp * XTS + c) = t;
    }
    for (int idx = tid; idx < (MCH * CIN) / 2; idx += 256) {
        v2f s = *(const v2f*)(wcmp + 2 * idx);
        v2bf t; t[0] = (__bf16)s[0]; t[1] = (__bf16)s[1];
        *(v2bf*)(wc + 2 * idx) = t;
    }
    // pre-zero y1t: out-of-image positions then stay zero (encoder padding)
    for (int idx = tid; idx < (325 * Y1S) / 8; idx += 256)
        *(v8bf*)(y1t + idx * 8) = (v8bf){};
    // encoder weights: coalesced read, scatter to tap-major K layout
    for (int idx = tid; idx < OENC * KENC; idx += 256) {
        int o = idx / KENC, r = idx - o * KENC;
        int m = r / 9, tap = r - m * 9;
        we[o * KENCP + tap * 48 + m] = (__bf16)wenc_g[idx];
    }
    for (int idx = tid; idx < OPAD * 16; idx += 256)             // K pad
        we[(idx >> 4) * KENCP + KENC + (idx & 15)] = (__bf16)0.f;
    for (int idx = tid; idx < (OPAD - OENC) * KENCP; idx += 256) // o pad rows
        we[OENC * KENCP + idx] = (__bf16)0.f;
    if (tid < MCH)  bc[tid] = bcmp[tid];
    if (tid < OPAD) be[tid] = (tid < OENC) ? benc_g[tid] : 0.f;
    __syncthreads();

    // ---------- Phase A: compress GEMM  y1t[324 x 48] = relu(Wc x + b)
    {
        const int row = lane & 15;
        const int hi8 = (lane < 16) ? 0 : 8;
        for (int pair = wv; pair < 63; pair += 8) {     // 3 mtiles x 21 ntiles
            int mt = pair / 21, nt = pair - mt * 21;
            const __bf16* arow = wc + (mt * 16 + row) * CIN;
            int  nn  = nt * 16 + row;                   // B column position
            bool okn = nn < P1;
            int  nc  = okn ? nn : 0;
            int  r1  = nc / H1, c1 = nc - r1 * H1;
            const __bf16* brow = xt + ((r1 + 1) * H2 + (c1 + 1)) * XTS;
            v8f acc = {};
            #pragma unroll
            for (int kk = 0; kk < 2; ++kk) {            // K = 64 = 2 x 32
                const int c0 = kk * 32 + hi8;
                v16bf av = cat16(*(const v8bf*)(arow + c0),
                                 *(const v8bf*)(arow + c0 + 16));
                v16bf bv = cat16(*(const v8bf*)(brow + c0),
                                 *(const v8bf*)(brow + c0 + 16));
                acc = __builtin_amdgcn_wmma_f32_16x16x32_bf16(
                          false, av, false, bv, (short)0, acc, false, false);
            }
            // select-free epilogue: vector bias load, address-redirect store
            int  ir = th0 + r1 - 1, ic = tw0 + c1 - 1;
            bool valid = okn && (unsigned)ir < (unsigned)HH
                             && (unsigned)ic < (unsigned)WWID;
            const float* bcp = bc + mt * 16 + hi8;
            v4f b0 = *(const v4f*)bcp;
            v4f b1 = *(const v4f*)(bcp + 4);
            v8bf outv;
            #pragma unroll
            for (int v = 0; v < 8; ++v) {
                float bias = (v < 4) ? b0[v] : b1[v - 4];
                outv[v] = (__bf16)fmaxf(acc[v] + bias, 0.f);
            }
            int rowp = valid ? nn : 324;    // invalid lanes -> dummy row; real
            *(v8bf*)(y1t + rowp * Y1S + mt * 16 + hi8) = outv; // row stays 0
        }
    }
    __syncthreads();

    // ---------- Phase B: encoder implicit GEMM, two interleaved WMMA chains
    // K tap-major: each 8-run stays in one tap for both lane halves, so tap
    // indices fold to constants under full unroll.
    {
        const int row = lane & 15;
        const int hi8 = (lane < 16) ? 0 : 8;
        for (int pr = wv; pr < 56; pr += 8) {           // 2 of 112 pairs / iter
            const int pa0 = pr, pa1 = pr + 56;
            const int ot0 = pa0 >> 4, nt0 = pa0 & 15;
            const int ot1 = pa1 >> 4, nt1 = pa1 & 15;
            const __bf16* arow0 = we + (ot0 * 16 + row) * KENCP;
            const __bf16* arow1 = we + (ot1 * 16 + row) * KENCP;
            const int p0 = nt0 * 16 + row, p1 = nt1 * 16 + row;
            int yof0[9], yof1[9];
            #pragma unroll
            for (int t = 0; t < 9; ++t) {
                const int dy = t / 3, dx = t % 3;
                yof0[t] = (((p0 >> 4) + dy) * H1 + (p0 & 15) + dx) * Y1S;
                yof1[t] = (((p1 >> 4) + dy) * H1 + (p1 & 15) + dx) * Y1S;
            }
            v8f acc0 = {}, acc1 = {};
            #pragma unroll
            for (int kk = 0; kk < 14; ++kk) {
                const int  kb = kk * 32;
                const int  t0 = kb / 48,        m0 = kb % 48;
                const int  t1 = (kb + 16) / 48, m1 = (kb + 16) % 48;
                const bool hiOK = (kb + 16) < KENC;      // folds per iteration

                v16bf av0 = cat16(*(const v8bf*)(arow0 + kb + hi8),
                                  *(const v8bf*)(arow0 + kb + hi8 + 16));
                v8bf  bl0 = *(const v8bf*)(y1t + yof0[t0] + m0 + hi8);
                v8bf  bh0 = {};
                if (hiOK) bh0 = *(const v8bf*)(y1t + yof0[t1] + m1 + hi8);
                acc0 = __builtin_amdgcn_wmma_f32_16x16x32_bf16(
                           false, av0, false, cat16(bl0, bh0),
                           (short)0, acc0, false, false);

                v16bf av1 = cat16(*(const v8bf*)(arow1 + kb + hi8),
                                  *(const v8bf*)(arow1 + kb + hi8 + 16));
                v8bf  bl1 = *(const v8bf*)(y1t + yof1[t0] + m0 + hi8);
                v8bf  bh1 = {};
                if (hiOK) bh1 = *(const v8bf*)(y1t + yof1[t1] + m1 + hi8);
                acc1 = __builtin_amdgcn_wmma_f32_16x16x32_bf16(
                           false, av1, false, cat16(bl1, bh1),
                           (short)0, acc1, false, false);
            }
            // scatter epilogue into [k*4+s] row layout (+bias); pad -> 100..115
            {
                const float* bep = be + ot0 * 16 + hi8;
                v4f e0 = *(const v4f*)bep, e1 = *(const v4f*)(bep + 4);
                #pragma unroll
                for (int v = 0; v < 8; ++v) {
                    int oo = ot0 * 16 + hi8 + v;
                    int s  = oo / 25, k = oo - s * 25;
                    int off = (oo < OENC) ? (k * 4 + s) : oo;   // scratch slots
                    float bias = (v < 4) ? e0[v] : e1[v - 4];
                    y2t[p0 * Y2S + off] = acc0[v] + bias;
                }
            }
            {
                const float* bep = be + ot1 * 16 + hi8;
                v4f e0 = *(const v4f*)bep, e1 = *(const v4f*)(bep + 4);
                #pragma unroll
                for (int v = 0; v < 8; ++v) {
                    int oo = ot1 * 16 + hi8 + v;
                    int s  = oo / 25, k = oo - s * 25;
                    int off = (oo < OENC) ? (k * 4 + s) : oo;
                    float bias = (v < 4) ? e0[v] : e1[v - 4];
                    y2t[p1 * Y2S + off] = acc1[v] + bias;
                }
            }
        }
    }
    __syncthreads();

    // ---------- Phase C: softmax over 25 taps, all 4 subpixels vectorized
    {
        float* rp = y2t + tid * Y2S;
        v4f mx = {-1e30f, -1e30f, -1e30f, -1e30f};
        for (int k = 0; k < 25; ++k) {
            v4f t = *(const v4f*)(rp + 4 * k);
            #pragma unroll
            for (int s = 0; s < 4; ++s) mx[s] = fmaxf(mx[s], t[s]);
        }
        v4f sum = {};
        for (int k = 0; k < 25; ++k) {
            v4f t = *(const v4f*)(rp + 4 * k);
            #pragma unroll
            for (int s = 0; s < 4; ++s) { t[s] = __expf(t[s] - mx[s]);
                                          sum[s] += t[s]; }
            *(v4f*)(rp + 4 * k) = t;
        }
        v4f inv;
        #pragma unroll
        for (int s = 0; s < 4; ++s) inv[s] = __builtin_amdgcn_rcpf(sum[s]);
        for (int k = 0; k < 25; ++k) {
            v4f t = *(const v4f*)(rp + 4 * k);
            #pragma unroll
            for (int s = 0; s < 4; ++s) t[s] *= inv[s];
            *(v4f*)(rp + 4 * k) = t;
        }
    }
    __syncthreads();

    // ---------- Phase D: weighted reassembly + PixelShuffle(2), float2 stores
    {
        const int p = tid;
        const int prow = p >> 4, pcol = p & 15;
        const int h = th0 + prow, w = tw0 + pcol;
        const float* wrow = y2t + p * Y2S;
        for (int cb = 0; cb < 8; ++cb) {                // 8 channels per pass
            float accd[4][8];
            #pragma unroll
            for (int s = 0; s < 4; ++s)
                #pragma unroll
                for (int j = 0; j < 8; ++j) accd[s][j] = 0.f;
            for (int k = 0; k < 25; ++k) {
                int dy = k / 5, dx = k - dy * 5;
                v8bf xv = *(const v8bf*)(xt + ((prow + dy) * H2 + (pcol + dx)) * XTS
                                            + cb * 8);
                v4f wg = *(const v4f*)(wrow + 4 * k);   // s = 0..3
                #pragma unroll
                for (int j = 0; j < 8; ++j) {
                    float xf = (float)xv[j];
                    accd[0][j] += wg[0] * xf;  accd[1][j] += wg[1] * xf;
                    accd[2][j] += wg[2] * xf;  accd[3][j] += wg[3] * xf;
                }
            }
            // ch = s*64 + c ; subpixel = ch&3 ; channel quads -> 2x float2 rows
            #pragma unroll
            for (int s = 0; s < 4; ++s) {
                #pragma unroll
                for (int q = 0; q < 2; ++q) {
                    int c  = cb * 8 + q * 4;
                    int co = (s * 64 + c) >> 2;
                    float* o0 = out + (((size_t)b * 64 + co) * (HH * SF) + SF * h)
                                          * (WWID * SF) + SF * w;
                    v2f t0 = {accd[s][q * 4 + 0], accd[s][q * 4 + 1]}; // row 2h
                    v2f t1 = {accd[s][q * 4 + 2], accd[s][q * 4 + 3]}; // row 2h+1
                    *(v2f*)o0 = t0;
                    *(v2f*)(o0 + WWID * SF) = t1;
                }
            }
        }
    }
}

extern "C" void kernel_launch(void* const* d_in, const int* in_sizes, int n_in,
                              void* d_out, int out_size, void* d_ws, size_t ws_size,
                              hipStream_t stream) {
    const float* x   = (const float*)d_in[0];
    const float* wcp = (const float*)d_in[1];
    const float* bcp = (const float*)d_in[2];
    const float* wen = (const float*)d_in[3];
    const float* ben = (const float*)d_in[4];
    float* out = (float*)d_out;

    hipFuncSetAttribute((const void*)carafe_fused,
                        hipFuncAttributeMaxDynamicSharedMemorySize, SMEM_BYTES);

    dim3 grid(WWID / TILE, HH / TILE, NB);   // 8 x 8 x 8 = 512 workgroups
    carafe_fused<<<grid, dim3(256), SMEM_BYTES, stream>>>(x, wcp, bcp, wen, ben, out);
}
